// SimpleTransformer_60756607369845
// MI455X (gfx1250) — matrix operand order repaired and checked by reading, not compiled, not run on hardware
//
#include <hip/hip_runtime.h>
#include <hip/hip_bf16.h>
#include <hip/hip_fp16.h>

typedef __attribute__((ext_vector_type(16))) _Float16 v16h;
typedef __attribute__((ext_vector_type(8)))  _Float16 v8h;
typedef __attribute__((ext_vector_type(8)))  float    v8f;

#define B_    2
#define NQ_   8192
#define NK_   8192
#define DIN_  256
#define DH_   256

// ---------------------------------------------------------------------------
// Projection: y = x @ W + b, output f16. One block per row (256 threads).
// transpose==1 stores y as [B, DH, rows_per_batch] (for V, so the attention
// kernel's PV B-operand is contiguous per lane).
// ---------------------------------------------------------------------------
__global__ __launch_bounds__(256) void proj_f16_kernel(
    const float* __restrict__ x, const float* __restrict__ W,
    const float* __restrict__ bias, _Float16* __restrict__ y,
    int transpose, int rows_per_batch)
{
    __shared__ float xs[DIN_];
    const int row = blockIdx.x;       // 0 .. B*N-1
    const int h   = threadIdx.x;      // 0 .. 255
    xs[h] = x[(size_t)row * DIN_ + h];
    __syncthreads();

    float acc = bias[h];
#pragma unroll 8
    for (int d = 0; d < DIN_; ++d)
        acc = fmaf(xs[d], W[(size_t)d * DH_ + h], acc);

    if (transpose) {
        const int b = row / rows_per_batch;
        const int n = row % rows_per_batch;
        y[((size_t)b * DH_ + h) * rows_per_batch + n] = (_Float16)acc;
    } else {
        y[(size_t)row * DH_ + h] = (_Float16)acc;
    }
}

// ---------------------------------------------------------------------------
// CDNA5 async copy helpers (ASYNCcnt-tracked VMEM -> LDS DMA).
// ---------------------------------------------------------------------------
__device__ __forceinline__ uint32_t lds_off(const void* p) {
    // generic pointer to LDS: low 32 bits are the LDS byte offset
    return (uint32_t)(uintptr_t)p;
}

__device__ __forceinline__ void async_b128(uint32_t ldsaddr, const void* gaddr) {
    asm volatile("global_load_async_to_lds_b128 %0, %1, off"
                 :: "v"(ldsaddr), "v"((uint64_t)(uintptr_t)gaddr)
                 : "memory");
}

__device__ __forceinline__ void wait_async0() {
    asm volatile("s_wait_asynccnt 0" ::: "memory");
}

// ---------------------------------------------------------------------------
// Flash attention, one wave32 per 16-query tile, 4 waves/block sharing
// double-buffered LDS K/V tiles filled by async-to-LDS DMA.
//   S(16x32) = Q(16x256) @ K^T          -> 16 x wmma_f32_16x16x32_f16
//   online softmax (shfl_xor within 16-lane halves, v_exp_f32)
//   O(16x256) += P(16x32) @ V(32x256)   -> 16 x wmma_f32_16x16x32_f16
// ---------------------------------------------------------------------------
__global__ __launch_bounds__(128, 1) void attn_flash_wmma_kernel(
    const _Float16* __restrict__ qe,   // [B, NQ, DH] row-major f16
    const _Float16* __restrict__ ke,   // [B, NK, DH] row-major f16
    const _Float16* __restrict__ vT,   // [B, DH, NK] f16 (transposed V)
    float* __restrict__ out)           // [B, NQ, DH] f32
{
    // double-buffered tiles + per-wave P transpose bounce (68 KB total)
    __shared__ __align__(64) _Float16 kbuf[2][32 * DH_];   // [j][d]
    __shared__ __align__(64) _Float16 vbuf[2][DH_ * 32];   // [h][j]
    __shared__ __align__(64) _Float16 pbuf[4][16 * 32];    // [m][j]

    const int tid  = threadIdx.x;
    const int wave = tid >> 5;
    const int lane = tid & 31;
    const int hi   = lane >> 4;
    const int lo   = lane & 15;

    const int tile = blockIdx.x * 4 + wave;
    const int b    = tile / (NQ_ / 16);
    const int qb   = (tile % (NQ_ / 16)) * 16;

    // cooperative async fill of one K tile + one V tile (32 KB / 128 thr)
    const _Float16* keb = ke + (size_t)b * NK_ * DH_;
    const _Float16* vTb = vT + (size_t)b * DH_ * NK_;
    const int kj   = tid >> 2;          // K: row 0..31
    const int kcol = (tid & 3) * 64;    // K: 128-byte sub-row
    auto fill_tiles = [&](int buf, int jb) {
        // K tile: rows jb..jb+31, 512 B each
        const _Float16* kg = keb + ((size_t)(jb + kj)) * DH_ + kcol;
        uint32_t kl = lds_off(&kbuf[buf][kj * DH_ + kcol]);
#pragma unroll
        for (int u = 0; u < 8; ++u)
            async_b128(kl + u * 16, kg + u * 8);
        // V tile: 256 rows of 64 B (two rows per thread)
        const _Float16* vg = vTb + ((size_t)(tid * 2)) * NK_ + jb;
        uint32_t vl = lds_off(&vbuf[buf][tid * 2 * 32]);
#pragma unroll
        for (int u = 0; u < 4; ++u) {
            async_b128(vl +      u * 16, vg           + u * 8);
            async_b128(vl + 64 + u * 16, vg + NK_     + u * 8);
        }
    };

    // ---- Q tile in A-operand layout, resident in VGPRs ------------------
    const _Float16* qrow = qe + ((size_t)b * NQ_ + qb + lo) * DH_;
    v16h aq[8];
#pragma unroll
    for (int c = 0; c < 8; ++c) {
        v8h q0 = *(const v8h*)(qrow + c * 32 + hi * 16);
        v8h q1 = *(const v8h*)(qrow + c * 32 + hi * 16 + 8);
        aq[c] = __builtin_shufflevector(q0, q1,
                 0,1,2,3,4,5,6,7,8,9,10,11,12,13,14,15);
    }

    const v8f zf = {};
    v8f o[16];
#pragma unroll
    for (int t = 0; t < 16; ++t) o[t] = zf;

    float mrow[8], lrow[8];
#pragma unroll
    for (int r = 0; r < 8; ++r) { mrow[r] = -3.0e38f; lrow[r] = 0.0f; }

    _Float16* pb = pbuf[wave];

    // prologue: fill buffer 0
    fill_tiles(0, 0);
    wait_async0();
    __syncthreads();

    for (int jb = 0; jb < NK_; jb += 32) {
        const int cur = (jb >> 5) & 1;

        // overlap: DMA next tile into the other buffer while we compute
        if (jb + 32 < NK_) fill_tiles(1 - cur, jb + 32);

        // ---- S = Q @ K^T over d=256 (from LDS) --------------------------
        const _Float16* k0 = &kbuf[cur][lo * DH_ + hi * 16];
        const _Float16* k1 = k0 + 16 * DH_;
        v8f s0 = zf, s1 = zf;
#pragma unroll
        for (int c = 0; c < 8; ++c) {
            v16h bk0 = *(const v16h*)(k0 + c * 32);
            v16h bk1 = *(const v16h*)(k1 + c * 32);
            s0 = __builtin_amdgcn_wmma_f32_16x16x32_f16(
                     false, aq[c], false, bk0, (short)0, s0, false, false);
            s1 = __builtin_amdgcn_wmma_f32_16x16x32_f16(
                     false, aq[c], false, bk1, (short)0, s1, false, false);
        }

        // ---- online softmax: lane holds rows m=r+8*hi, col lo -----------
        float corr[8];
#pragma unroll
        for (int r = 0; r < 8; ++r) {
            float mx = fmaxf(s0[r], s1[r]);
            mx = fmaxf(mx, __shfl_xor(mx, 1));
            mx = fmaxf(mx, __shfl_xor(mx, 2));
            mx = fmaxf(mx, __shfl_xor(mx, 4));
            mx = fmaxf(mx, __shfl_xor(mx, 8));
            const float mnew = fmaxf(mrow[r], mx);
            const float p0 = __expf(s0[r] - mnew);
            const float p1 = __expf(s1[r] - mnew);
            corr[r] = __expf(mrow[r] - mnew);
            mrow[r] = mnew;
            float ls = p0 + p1;
            ls += __shfl_xor(ls, 1);
            ls += __shfl_xor(ls, 2);
            ls += __shfl_xor(ls, 4);
            ls += __shfl_xor(ls, 8);
            lrow[r] = lrow[r] * corr[r] + ls;

            const int m = r + 8 * hi;
            pb[m * 32 +      lo] = (_Float16)p0;
            pb[m * 32 + 16 + lo] = (_Float16)p1;
        }

        // wave-local LDS store->load ordering for the P transpose bounce
        asm volatile("s_wait_dscnt 0" ::: "memory");

        v8h p0v = *(const v8h*)(pb + lo * 32 + hi * 16);
        v8h p1v = *(const v8h*)(pb + lo * 32 + hi * 16 + 8);
        v16h ap = __builtin_shufflevector(p0v, p1v,
                   0,1,2,3,4,5,6,7,8,9,10,11,12,13,14,15);

        // ---- rescale accumulator, then O += P @ V (V from LDS) ----------
#pragma unroll
        for (int t = 0; t < 16; ++t) {
#pragma unroll
            for (int r = 0; r < 8; ++r) o[t][r] *= corr[r];
        }
#pragma unroll
        for (int t = 0; t < 16; ++t) {
            v16h bv = *(const v16h*)(&vbuf[cur][(t * 16 + lo) * 32 + hi * 16]);
            o[t] = __builtin_amdgcn_wmma_f32_16x16x32_f16(
                       false, ap, false, bv, (short)0, o[t], false, false);
        }

        // next tile's DMA must be complete and everyone done reading `cur`
        wait_async0();
        __syncthreads();
    }

    // ---- normalize and store f32 output --------------------------------
    float inv[8];
#pragma unroll
    for (int r = 0; r < 8; ++r) inv[r] = 1.0f / lrow[r];
#pragma unroll
    for (int t = 0; t < 16; ++t) {
#pragma unroll
        for (int r = 0; r < 8; ++r) {
            out[((size_t)b * NQ_ + qb + r + 8 * hi) * DH_ + t * 16 + lo] =
                o[t][r] * inv[r];
        }
    }
}

// ---------------------------------------------------------------------------
extern "C" void kernel_launch(void* const* d_in, const int* in_sizes, int n_in,
                              void* d_out, int out_size, void* d_ws, size_t ws_size,
                              hipStream_t stream) {
    const float* q  = (const float*)d_in[0];
    const float* k  = (const float*)d_in[1];
    const float* v  = (const float*)d_in[2];
    const float* Wq = (const float*)d_in[3];
    const float* bq = (const float*)d_in[4];
    const float* Wk = (const float*)d_in[5];
    const float* bk = (const float*)d_in[6];
    const float* Wv = (const float*)d_in[7];
    const float* bv = (const float*)d_in[8];
    // d_in[9] = chunk_size: reference-side memory optimization, ignored.

    float* out = (float*)d_out;

    char* ws = (char*)d_ws;
    const size_t emb_bytes = (size_t)B_ * NQ_ * DH_ * sizeof(_Float16); // 8 MB
    _Float16* qe = (_Float16*)(ws);
    _Float16* ke = (_Float16*)(ws + emb_bytes);
    _Float16* vT = (_Float16*)(ws + 2 * emb_bytes);

    proj_f16_kernel<<<dim3(B_ * NQ_), 256, 0, stream>>>(q, Wq, bq, qe, 0, NQ_);
    proj_f16_kernel<<<dim3(B_ * NK_), 256, 0, stream>>>(k, Wk, bk, ke, 0, NK_);
    proj_f16_kernel<<<dim3(B_ * NK_), 256, 0, stream>>>(v, Wv, bv, vT, 1, NK_);

    // B*NQ/16 = 1024 query tiles, 4 waves (tiles) per 128-thread block;
    // blocks are 4-tile aligned so all waves of a block share batch b.
    attn_flash_wmma_kernel<<<dim3(B_ * NQ_ / 16 / 4), 128, 0, stream>>>(
        qe, ke, vT, out);
}
